// HyperConv_68994354642978
// MI455X (gfx1250) — matrix operand order repaired and checked by reading, not compiled, not run on hardware
//
#include <hip/hip_runtime.h>
#include <hip/hip_bf16.h>
#include <stdint.h>

// ---------------------------------------------------------------------------
// HyperConv for MI455X (gfx1250): f16 WMMA implicit-GEMM 3D conv with
// TDM-staged LDS halo tiles.
//   BS=4, IC=16, OC=16, KS=3 (27 taps), D=96, HYP=128, pad=1.
// Stage 1: hypernet GEMM -> f16 weights pre-swizzled into WMMA A-fragment
//          layout (2 taps per K=32 fragment; tap 27 zero-padded) + f32 bias.
// Stage 2: x (f32 NCDHW) -> padded channels-last f16 [b][98][98][98][16]
//          (120 MB, L2-resident).
// Stage 3: each block computes a 16x4x2-voxel x 16-oc tile. Wave 0 DMAs the
//          18x6x4-voxel x 16-ch halo box (13.8 KB) into LDS via the Tensor
//          Data Mover (tensor_load_to_lds + s_wait_tensorcnt), then each of
//          the 8 waves runs 14 v_wmma_f32_16x16x32_f16 off ds_load_b128
//          B-fragments. Cuts L2 B-traffic ~8x vs direct global reads.
// ---------------------------------------------------------------------------

typedef __attribute__((ext_vector_type(16))) _Float16 v16h;
typedef __attribute__((ext_vector_type(8)))  float    v8f;
typedef __attribute__((ext_vector_type(4)))  unsigned int v4u;
typedef __attribute__((ext_vector_type(8)))  int      v8i;
typedef __attribute__((ext_vector_type(4)))  int      v4i;

#define D96   96
#define DP    98            // padded dim
#define NTAP  27
#define NPAIR 14            // ceil(27/2), tap 27 is zero weight
#define HYP   128

// conv tiling: per block 16(x) x 4(y) x 2(z) out voxels, 8 waves (1 xy-row ea)
#define TILE_Y 4
#define TILE_Z 2
#define HALO_X 18
#define HALO_Y 6
#define HALO_Z 4
#define HALO_VOX (HALO_X * HALO_Y * HALO_Z)   // 432 voxels * 32B = 13824 B LDS

// ws layout (bytes):
//   [0, 57344)          : A fragments, half[4][14][32][16]
//   [57344, 57600)      : bias, float[4][16]
//   [61440, 61440+120MB): padded f16 x, half[4][98][98][98][16]
#define WS_BIAS_OFF  57344
#define WS_XPAD_OFF  61440

// ---------------------------------------------------------------------------
// Kernel 1: hypernetwork. Threads [0,28672) produce one A-fragment half each,
// written at its exact WMMA lane/half slot. Threads [28672,28736) do bias.
// ---------------------------------------------------------------------------
__global__ __launch_bounds__(256)
void hyper_k(const float* __restrict__ hyp, const float* __restrict__ Wk,
             const float* __restrict__ bk,  const float* __restrict__ Wb,
             const float* __restrict__ bb,
             _Float16* __restrict__ Afrag, float* __restrict__ bias)
{
    int t = blockIdx.x * 256 + threadIdx.x;
    if (t < 4 * NPAIR * 32 * 16) {
        // t = ((b*14 + p)*32 + lane)*16 + h   (storage == decode)
        int h    = t & 15;
        int lane = (t >> 4) & 31;
        int rest = t >> 9;
        int p    = rest % NPAIR;
        int b    = rest / NPAIR;
        // 16-bit A 16x32 layout (contiguous k): M = lane&15,
        // K = h + 16*(lane>>4)
        int M   = lane & 15;
        int K   = h + ((lane >> 4) << 4);
        int sel = K >> 4;        // which tap of the pair
        int ic  = K & 15;
        int tap = 2 * p + sel;
        float w = 0.0f;
        if (tap < NTAP) {
            int j = (M * 16 + ic) * NTAP + tap;   // reshape [OC][IC][27]
            const float* hrow = hyp + b * HYP;
            const float* wrow = Wk + (size_t)j * HYP;
            float acc = bk[j];
            #pragma unroll 8
            for (int q = 0; q < HYP; ++q) acc += hrow[q] * wrow[q];
            w = acc;
        }
        Afrag[t] = (_Float16)w;
    } else if (t < 4 * NPAIR * 32 * 16 + 64) {
        int u  = t - 4 * NPAIR * 32 * 16;
        int b  = u >> 4;
        int oc = u & 15;
        const float* hrow = hyp + b * HYP;
        const float* wrow = Wb + oc * HYP;
        float acc = bb[oc];
        #pragma unroll 8
        for (int q = 0; q < HYP; ++q) acc += hrow[q] * wrow[q];
        bias[u] = acc;
    }
}

// ---------------------------------------------------------------------------
// Kernel 2: x f32 NCDHW -> zero-padded channels-last f16 [b][zp][yp][xp][ic].
// ---------------------------------------------------------------------------
__global__ __launch_bounds__(256)
void padcvt_k(const float* __restrict__ x, _Float16* __restrict__ xpad)
{
    size_t t = (size_t)blockIdx.x * 256 + threadIdx.x;
    const size_t NV = (size_t)4 * DP * DP * DP;
    if (t >= NV) return;
    int xi = (int)(t % DP);
    size_t r = t / DP;
    int yi = (int)(r % DP); r /= DP;
    int zi = (int)(r % DP);
    int b  = (int)(r / DP);

    v16h v = {};   // zero halo
    if (xi > 0 && xi < DP - 1 && yi > 0 && yi < DP - 1 && zi > 0 && zi < DP - 1) {
        const size_t CSTR = (size_t)D96 * D96 * D96;   // per-channel stride
        size_t base = ((((size_t)b * 16) * D96 + (zi - 1)) * D96 + (yi - 1)) * D96
                      + (xi - 1);
        #pragma unroll
        for (int ic = 0; ic < 16; ++ic)
            v[ic] = (_Float16)x[base + (size_t)ic * CSTR];
    }
    *(v16h*)(xpad + t * 16) = v;
}

// ---------------------------------------------------------------------------
// Kernel 3: WMMA conv with LDS halo tile.
// Block = (b, z0..z0+1, y0..y0+3, 16 x-voxels); wave w -> (zz=w>>2, yy=w&3).
// ---------------------------------------------------------------------------
__global__ __launch_bounds__(256)
void conv_k(const _Float16* __restrict__ xpad, const _Float16* __restrict__ Afrag,
            const float* __restrict__ bias, float* __restrict__ out)
{
    __shared__ __align__(32) _Float16 lds[HALO_VOX * 16];

    int tid  = threadIdx.x;
    int lane = tid & 31;
    int wave = tid >> 5;

    int bid = blockIdx.x;                 // 6 * 24 * 48 * 4 = 27648 blocks
    int xb = bid % 6;
    int yb = (bid / 6) % 24;
    int zb = (bid / 144) % 48;
    int b  = bid / 6912;
    int x0g = xb * 16;                    // tile origin in padded coords
    int y0  = yb * TILE_Y;
    int z0  = zb * TILE_Z;

    // ---- stage halo box [z0..z0+3][y0..y0+5][x0g..x0g+17][0..15] into LDS ----
    const size_t tile_origin =
        ((((size_t)b * DP + z0) * DP + y0) * DP + x0g) * 16;

#if defined(__has_builtin) && __has_builtin(__builtin_amdgcn_tensor_load_to_lds)
    if (wave == 0) {
        // ---- Tensor DMA descriptor (D#), per CDNA5 ISA 8.3-8.6 ----
        unsigned lds_off = (unsigned)(uintptr_t)(&lds[0]);   // low 32b = LDS offset
        unsigned long long gaddr =
            (unsigned long long)(uintptr_t)(xpad + tile_origin);

        const unsigned TD0 = 1u << 30, TD1 = 1u << 30, TD2 = 1u << 30; // never OOB
        const unsigned TILE0 = HALO_X * 16;          // 288 f16, contiguous x*ic
        const unsigned TILE1 = HALO_Y;               // 6 rows
        const unsigned TILE2 = HALO_Z;               // 4 planes
        const unsigned long long STR0 = (unsigned long long)DP * 16;      // 1568
        const unsigned long long STR1 = (unsigned long long)DP * DP * 16; // 153664

        v4u g0;
        g0[0] = 0x1u;                                        // count=1, user mode
        g0[1] = lds_off;                                     // lds_addr
        g0[2] = (unsigned)(gaddr & 0xFFFFFFFFu);             // global_addr[31:0]
        g0[3] = (unsigned)((gaddr >> 32) & 0x01FFFFFFu)      // global_addr[56:32]
                | (2u << 30);                                // type=2 ("image")

        v8i g1;
        g1[0] = (int)0x00010000u;                            // data_size=1 (2B)
        g1[1] = (int)((TD0 & 0xFFFFu) << 16);                // tensor_dim0[15:0]
        g1[2] = (int)(((TD0 >> 16) & 0xFFFFu) | ((TD1 & 0xFFFFu) << 16));
        g1[3] = (int)(((TD1 >> 16) & 0xFFFFu) | (TILE0 << 16));
        g1[4] = (int)((TILE1 & 0xFFFFu) | (TILE2 << 16));
        g1[5] = (int)(STR0 & 0xFFFFFFFFu);
        g1[6] = (int)(((STR0 >> 32) & 0xFFFFu) | ((STR1 & 0xFFFFu) << 16));
        g1[7] = (int)((STR1 >> 16) & 0xFFFFFFFFu);

        v4i g2;
        g2[0] = (int)TD2;   // tensor_dim2
        g2[1] = 0;          // tensor_dim3 (iterate disabled)
        g2[2] = 0;          // tensor_dim2_stride low (tile_dim3 unused)
        g2[3] = 0;          // tensor_dim2_stride hi | tile_dim3=0
        v4i g3 = {0, 0, 0, 0};
        v8i gx = {0, 0, 0, 0, 0, 0, 0, 0};   // extra operand of 6-arg builtin

        __builtin_amdgcn_tensor_load_to_lds(g0, g1, g2, g3, gx, 0);
        __builtin_amdgcn_s_wait_tensorcnt(0);
    }
#else
    // Fallback: cooperative copy, 432 voxel-vectors across 256 threads.
    for (int v = tid; v < HALO_VOX; v += 256) {
        int hx = v % HALO_X;
        int hy = (v / HALO_X) % HALO_Y;
        int hz = v / (HALO_X * HALO_Y);
        const _Float16* src = xpad +
            ((((size_t)b * DP + (z0 + hz)) * DP + (y0 + hy)) * DP + (x0g + hx)) * 16;
        *(v16h*)(&lds[(size_t)v * 16]) = *(const v16h*)src;
    }
#endif
    __syncthreads();

    // ---- compute: one (zz, yy) row of 16 voxels x 16 oc per wave ----
    int zz  = wave >> 2;       // 0..1
    int yy  = wave & 3;        // 0..3
    int n   = lane & 15;       // voxel column N
    int sel = lane >> 4;       // 0: first tap of pair, 1: second tap

    const char* lp = (const char*)lds;
    int lbase = ((zz * HALO_Y + yy) * HALO_X + n) * 32;   // bytes

    v8f acc = {};
    #pragma unroll
    for (int p = 0; p < NPAIR; ++p) {
        const int tA = 2 * p;
        const int tB = (2 * p + 1 > 26) ? 26 : 2 * p + 1;   // dup; weights are 0
        const int dA = (((tA / 9) * HALO_Y + (tA / 3) % 3) * HALO_X + (tA % 3)) * 32;
        const int dB = (((tB / 9) * HALO_Y + (tB / 3) % 3) * HALO_X + (tB % 3)) * 32;
        int d = sel ? dB : dA;                               // one v_cndmask
        v16h bf = *(const v16h*)(lp + lbase + d);            // 2x ds_load_b128
        v16h af = *(const v16h*)(Afrag + (((size_t)b * NPAIR + p) * 32 + lane) * 16);
        acc = __builtin_amdgcn_wmma_f32_16x16x32_f16(
                  false, af, false, bf, (short)0, acc, false, false);
    }

    // bias: acc[r] is oc = r + 8*sel, voxel x = x0g + n
    v8f bv = *(const v8f*)(bias + b * 16 + sel * 8);
    acc += bv;

    int z = z0 + zz, y = y0 + yy;
    #pragma unroll
    for (int rr = 0; rr < 8; ++rr) {
        int oc = rr + 8 * sel;
        out[((((size_t)(b * 16 + oc)) * D96 + z) * D96 + y) * D96 + x0g + n]
            = acc[rr];
    }
}

// ---------------------------------------------------------------------------
extern "C" void kernel_launch(void* const* d_in, const int* in_sizes, int n_in,
                              void* d_out, int out_size, void* d_ws, size_t ws_size,
                              hipStream_t stream)
{
    const float* x   = (const float*)d_in[0];
    const float* hyp = (const float*)d_in[1];
    const float* Wk  = (const float*)d_in[2];
    const float* bk  = (const float*)d_in[3];
    const float* Wb  = (const float*)d_in[4];
    const float* bb  = (const float*)d_in[5];
    float* out = (float*)d_out;

    _Float16* Afrag = (_Float16*)d_ws;
    float*    bias  = (float*)((char*)d_ws + WS_BIAS_OFF);
    _Float16* xpad  = (_Float16*)((char*)d_ws + WS_XPAD_OFF);

    // Stage 1: hypernet (28672 A-halves + 64 biases)
    hyper_k<<<(28736 + 255) / 256, 256, 0, stream>>>(hyp, Wk, bk, Wb, bb, Afrag, bias);

    // Stage 2: pad + f32->f16 transpose (4*98^3 voxels)
    const size_t NV = (size_t)4 * DP * DP * DP;
    padcvt_k<<<(unsigned)((NV + 255) / 256), 256, 0, stream>>>(x, xpad);

    // Stage 3: WMMA conv: 27648 blocks of 8 waves (16x4x2 voxel tile each)
    conv_k<<<27648, 256, 0, stream>>>(xpad, Afrag, bias, out);
}
